// UniversalCalculator_74380243632185
// MI455X (gfx1250) — compile-verified, matrix operand order checked by reading, not run on hardware
//
#include <hip/hip_runtime.h>

// ---------------- problem constants ----------------
#define E_   8
#define D_   2048
#define H_   688
#define HP_  704          // H padded to multiple of 32 for GEMM2 K-dim
#define T_   4096
#define K_   2
#define CAP_ 1536
#define NFLAT (T_*K_)     // 8192

typedef __bf16 bf16;
typedef __bf16 v8bf  __attribute__((ext_vector_type(8)));
typedef __bf16 v16bf __attribute__((ext_vector_type(16)));
typedef float  v8f   __attribute__((ext_vector_type(8)));

// ---------------- helpers ----------------
__global__ void fill_i32_kernel(int* p, int v, int n) {
    int i = blockIdx.x * blockDim.x + threadIdx.x;
    if (i < n) p[i] = v;
}

// Stable per-expert ranking identical to a stable argsort by expert id.
// 1 block, 256 threads, chunk of 32 flat entries each.
__global__ void dispatch_kernel(const int* __restrict__ topk_idx,
                                int* __restrict__ counts,
                                int* __restrict__ slot,
                                int* __restrict__ slot2flat) {
    __shared__ int lc[256 * E_];
    const int th = threadIdx.x;
    const int base_i = th * (NFLAT / 256);

    int c[E_];
#pragma unroll
    for (int e = 0; e < E_; ++e) c[e] = 0;
    for (int j = 0; j < NFLAT / 256; ++j) {
        int e = topk_idx[base_i + j];
        c[e]++;
    }
#pragma unroll
    for (int e = 0; e < E_; ++e) lc[th * E_ + e] = c[e];
    __syncthreads();

    if (th < E_) {               // per-expert exclusive scan across chunks
        int run = 0;
        for (int t = 0; t < 256; ++t) {
            int v = lc[t * E_ + th];
            lc[t * E_ + th] = run;
            run += v;
        }
        counts[th] = run;
    }
    __syncthreads();

#pragma unroll
    for (int e = 0; e < E_; ++e) c[e] = lc[th * E_ + e];
    for (int j = 0; j < NFLAT / 256; ++j) {
        int i = base_i + j;
        int e = topk_idx[i];
        int pos = c[e]++;
        if (pos < CAP_) {
            int s = e * CAP_ + pos;
            slot[i] = s;
            slot2flat[s] = i;
        } else {
            slot[i] = -1;        // dropped (never happens with this CAP)
        }
    }
}

// xb[s][d] = bf16(x[tok][d]) or 0 for empty slots.  grid.x = E*CAP
__global__ void gather_kernel(const float* __restrict__ x,
                              const int* __restrict__ slot2flat,
                              bf16* __restrict__ xb) {
    const int s = blockIdx.x;
    const int f = slot2flat[s];
    bf16* dst = xb + (size_t)s * D_;
    if (f >= 0) {
        const float* src = x + (size_t)(f / K_) * D_;
        for (int d = threadIdx.x; d < D_; d += blockDim.x)
            dst[d] = (bf16)src[d];
    } else {
        for (int d = threadIdx.x; d < D_; d += blockDim.x)
            dst[d] = (bf16)0.0f;
    }
}

// (E,D,H) f32 -> (E,H,D) bf16   (B-operand layout: N-major rows, K contiguous)
__global__ void transpose_dh_kernel(const float* __restrict__ w, bf16* __restrict__ wt) {
    const size_t total = (size_t)E_ * H_ * D_;
    size_t o = (size_t)blockIdx.x * blockDim.x + threadIdx.x;
    if (o >= total) return;
    int d = (int)(o % D_);
    int h = (int)((o / D_) % H_);
    int e = (int)(o / ((size_t)D_ * H_));
    wt[o] = (bf16)w[((size_t)e * D_ + d) * H_ + h];
}

// (E,H,D) f32 -> (E,D,HP) bf16, zero pad K 688->704
__global__ void transpose_hd_pad_kernel(const float* __restrict__ w, bf16* __restrict__ wt) {
    const size_t total = (size_t)E_ * D_ * HP_;
    size_t o = (size_t)blockIdx.x * blockDim.x + threadIdx.x;
    if (o >= total) return;
    int hp = (int)(o % HP_);
    int d  = (int)((o / HP_) % D_);
    int e  = (int)(o / ((size_t)HP_ * D_));
    wt[o] = (hp < H_) ? (bf16)w[((size_t)e * H_ + hp) * D_ + d] : (bf16)0.0f;
}

// ---- WMMA fragment load (A or B operand; 16 rows x 32 K, bf16) ----
// lane<16 : row=lane,    K = [k0..k0+7] ++ [k0+16..k0+23]
// lane>=16: row=lane-16, K = [k0+8..k0+15] ++ [k0+24..k0+31]
__device__ __forceinline__ const bf16* frag_ptr(const bf16* __restrict__ base,
                                                int row0, int ld, int k0, int lane) {
    return base + (size_t)(row0 + (lane & 15)) * ld + k0 + ((lane & 16) ? 8 : 0);
}

__device__ __forceinline__ v16bf load_frag(const bf16* __restrict__ base,
                                           int row0, int ld, int k0, int lane) {
    const bf16* p = frag_ptr(base, row0, ld, k0, lane);
    v8bf lo = *(const v8bf*)(p);
    v8bf hi = *(const v8bf*)(p + 16);
    return __builtin_shufflevector(lo, hi, 0,1,2,3,4,5,6,7,8,9,10,11,12,13,14,15);
}

// GEMM1 fused: G = xb@Wg, U = xb@Wu, h = silu(G)*U  (bf16 out, K-padded rows)
// block = 128 (4 waves). wave w -> n-tile blockIdx.y*4+w. blockIdx.x -> 64-row strip.
// Branch-free k-loop: the one out-of-range wave computes a clamped duplicate
// tile (L0 hits) and skips only its stores.
__global__ void gemm1_glu_kernel(const bf16* __restrict__ xb,
                                 const bf16* __restrict__ wgt,   // (E,H,D)
                                 const bf16* __restrict__ wut,   // (E,H,D)
                                 const int* __restrict__ counts,
                                 bf16* __restrict__ h) {
    const int e = blockIdx.z;
    int cnt = counts[e]; if (cnt > CAP_) cnt = CAP_;
    const int m0 = blockIdx.x * 64;
    if (m0 >= cnt) return;                      // skip empty capacity (block-uniform)
    const int wave = threadIdx.x >> 5;
    const int lane = threadIdx.x & 31;
    const int n0 = (blockIdx.y * 4 + wave) * 16;
    const int n0c = (n0 < H_) ? n0 : (H_ - 16);

    const bf16* A  = xb  + (size_t)e * CAP_ * D_;
    const bf16* Bg = wgt + (size_t)e * H_ * D_;
    const bf16* Bu = wut + (size_t)e * H_ * D_;

    v8f cg[4], cu[4];
#pragma unroll
    for (int i = 0; i < 4; ++i) { cg[i] = (v8f){}; cu[i] = (v8f){}; }

    for (int k = 0; k < D_; k += 32) {
        if (k + 64 < D_)   // prefetch A stream 2 k-steps ahead (covers 16 rows)
            __builtin_prefetch(frag_ptr(A, m0, D_, k + 64, lane), 0, 0);
        v16bf bg = load_frag(Bg, n0c, D_, k, lane);
        v16bf bu = load_frag(Bu, n0c, D_, k, lane);
#pragma unroll
        for (int ms = 0; ms < 4; ++ms) {
            v16bf a = load_frag(A, m0 + ms * 16, D_, k, lane);
            cg[ms] = __builtin_amdgcn_wmma_f32_16x16x32_bf16(false, a, false, bg,
                                                             (short)0, cg[ms], false, false);
            cu[ms] = __builtin_amdgcn_wmma_f32_16x16x32_bf16(false, a, false, bu,
                                                             (short)0, cu[ms], false, false);
        }
    }

    if (n0 < H_) {
        // epilogue: h = g*sigmoid(g)*u  -> bf16, row-padded to HP_
        const int n = n0 + (lane & 15);
        const int mhi = (lane & 16) ? 8 : 0;
        bf16* Hrow = h + (size_t)e * CAP_ * HP_;
#pragma unroll
        for (int ms = 0; ms < 4; ++ms) {
#pragma unroll
            for (int j = 0; j < 8; ++j) {
                int m = m0 + ms * 16 + mhi + j;
                float g = cg[ms][j], u = cu[ms][j];
                float hv = g * u / (1.0f + __expf(-g));
                Hrow[(size_t)m * HP_ + n] = (bf16)hv;
            }
        }
    }
}

// GEMM2: yb = h @ Wd   (K=HP_, N=D_), f32 out.
// Each wave computes a 64x32 tile (two 16-wide n-tiles sharing the A fragments)
// -> 8 WMMAs per 12 b128 loads per k-step.
__global__ void gemm2_down_kernel(const bf16* __restrict__ h,
                                  const bf16* __restrict__ wdt,  // (E,D,HP)
                                  const int* __restrict__ counts,
                                  float* __restrict__ yb) {
    const int e = blockIdx.z;
    int cnt = counts[e]; if (cnt > CAP_) cnt = CAP_;
    const int m0 = blockIdx.x * 64;
    if (m0 >= cnt) return;
    const int wave = threadIdx.x >> 5;
    const int lane = threadIdx.x & 31;
    const int n0 = blockIdx.y * 128 + wave * 32;   // two tiles: n0, n0+16

    const bf16* A = h   + (size_t)e * CAP_ * HP_;
    const bf16* B = wdt + (size_t)e * D_ * HP_;

    v8f cy0[4], cy1[4];
#pragma unroll
    for (int i = 0; i < 4; ++i) { cy0[i] = (v8f){}; cy1[i] = (v8f){}; }

    for (int k = 0; k < HP_; k += 32) {
        if (k + 64 < HP_)
            __builtin_prefetch(frag_ptr(A, m0, HP_, k + 64, lane), 0, 0);
        v16bf b0 = load_frag(B, n0,      HP_, k, lane);
        v16bf b1 = load_frag(B, n0 + 16, HP_, k, lane);
#pragma unroll
        for (int ms = 0; ms < 4; ++ms) {
            v16bf a = load_frag(A, m0 + ms * 16, HP_, k, lane);
            cy0[ms] = __builtin_amdgcn_wmma_f32_16x16x32_bf16(false, a, false, b0,
                                                              (short)0, cy0[ms], false, false);
            cy1[ms] = __builtin_amdgcn_wmma_f32_16x16x32_bf16(false, a, false, b1,
                                                              (short)0, cy1[ms], false, false);
        }
    }

    const int nl = lane & 15;
    const int mhi = (lane & 16) ? 8 : 0;
    float* Y = yb + (size_t)e * CAP_ * D_;
#pragma unroll
    for (int ms = 0; ms < 4; ++ms)
#pragma unroll
        for (int j = 0; j < 8; ++j) {
            int m = m0 + ms * 16 + mhi + j;
            Y[(size_t)m * D_ + n0 + nl]      = cy0[ms][j];
            Y[(size_t)m * D_ + n0 + 16 + nl] = cy1[ms][j];
        }
}

// y[t][d] = sum_k score[t,k] * yb[slot[t*K+k]][d]   (deterministic, no atomics)
__global__ void combine_kernel(const float* __restrict__ scores,
                               const int* __restrict__ slot,
                               const float* __restrict__ yb,
                               float* __restrict__ y) {
    const int t = blockIdx.x;
    const int d = blockIdx.y * blockDim.x + threadIdx.x;
    float acc = 0.0f;
#pragma unroll
    for (int k = 0; k < K_; ++k) {
        int i = t * K_ + k;
        int s = slot[i];
        if (s >= 0) acc += scores[i] * yb[(size_t)s * D_ + d];
    }
    y[(size_t)t * D_ + d] = acc;
}

// ---------------- host ----------------
extern "C" void kernel_launch(void* const* d_in, const int* in_sizes, int n_in,
                              void* d_out, int out_size, void* d_ws, size_t ws_size,
                              hipStream_t stream) {
    const float* x       = (const float*)d_in[0];
    const int*   topkIdx = (const int*)  d_in[1];
    const float* topkSc  = (const float*)d_in[2];
    const float* Wg      = (const float*)d_in[3];
    const float* Wu      = (const float*)d_in[4];
    const float* Wd      = (const float*)d_in[5];
    float* y = (float*)d_out;

    // workspace carve-up (256B aligned)
    char* ws = (char*)d_ws;
    size_t off = 0;
    auto alloc = [&](size_t bytes) -> char* {
        char* p = ws + off;
        off = (off + bytes + 255) & ~(size_t)255;
        return p;
    };
    int*  counts    = (int*) alloc(E_ * sizeof(int));
    int*  slot      = (int*) alloc((size_t)NFLAT * sizeof(int));
    int*  slot2flat = (int*) alloc((size_t)E_ * CAP_ * sizeof(int));
    bf16* wgt       = (bf16*)alloc((size_t)E_ * H_ * D_ * sizeof(bf16));
    bf16* wut       = (bf16*)alloc((size_t)E_ * H_ * D_ * sizeof(bf16));
    bf16* wdt       = (bf16*)alloc((size_t)E_ * D_ * HP_ * sizeof(bf16));
    bf16* xb        = (bf16*)alloc((size_t)E_ * CAP_ * D_ * sizeof(bf16));
    bf16* hbuf      = (bf16*)alloc((size_t)E_ * CAP_ * HP_ * sizeof(bf16));
    float* yb       = (float*)alloc((size_t)E_ * CAP_ * D_ * sizeof(float));
    (void)ws_size; (void)in_sizes; (void)n_in; (void)out_size;

    // 1) dispatch
    fill_i32_kernel<<<(E_ * CAP_ + 255) / 256, 256, 0, stream>>>(slot2flat, -1, E_ * CAP_);
    dispatch_kernel<<<1, 256, 0, stream>>>(topkIdx, counts, slot, slot2flat);

    // 2) gather + weight layout conversion (f32 -> bf16, B-operand layout)
    gather_kernel<<<E_ * CAP_, 256, 0, stream>>>(x, slot2flat, xb);
    {
        size_t tot = (size_t)E_ * H_ * D_;
        int blocks = (int)((tot + 255) / 256);
        transpose_dh_kernel<<<blocks, 256, 0, stream>>>(Wg, wgt);
        transpose_dh_kernel<<<blocks, 256, 0, stream>>>(Wu, wut);
    }
    {
        size_t tot = (size_t)E_ * D_ * HP_;
        int blocks = (int)((tot + 255) / 256);
        transpose_hd_pad_kernel<<<blocks, 256, 0, stream>>>(Wd, wdt);
    }

    // 3) fused gate/up GEMM + SiLU*mul  (grid: m-strips x n-blocks x experts)
    {
        dim3 grid(CAP_ / 64, (H_ / 16 + 3) / 4, E_);   // 24 x 11 x 8
        gemm1_glu_kernel<<<grid, 128, 0, stream>>>(xb, wgt, wut, counts, hbuf);
    }
    // 4) down GEMM (64x128 block tile, 64x32 per wave)
    {
        dim3 grid(CAP_ / 64, D_ / 128, E_);            // 24 x 16 x 8
        gemm2_down_kernel<<<grid, 128, 0, stream>>>(hbuf, wdt, counts, yb);
    }
    // 5) combine
    {
        dim3 grid(T_, D_ / 256);
        combine_kernel<<<grid, 256, 0, stream>>>(topkSc, slot, yb, y);
    }
}
